// LinearAttentionModel_8624294330487
// MI455X (gfx1250) — matrix operand order repaired and checked by reading, not compiled, not run on hardware
//
#include <hip/hip_runtime.h>

// ---------------------------------------------------------------------------
// Types for CDNA5 WMMA (gfx1250, wave32)
// ---------------------------------------------------------------------------
typedef __bf16 bf16_t;
typedef __attribute__((ext_vector_type(16))) __bf16 v16bf;
typedef __attribute__((ext_vector_type(8)))  __bf16 v8bf;
typedef __attribute__((ext_vector_type(8)))  float  v8f;

static __device__ __forceinline__ float silu_f(float x)    { return x / (1.0f + __expf(-x)); }
static __device__ __forceinline__ float sigmoid_f(float x) { return 1.0f / (1.0f + __expf(-x)); }

static __device__ __forceinline__ v8f wmma_bf16(v16bf a, v16bf b, v8f c) {
  return __builtin_amdgcn_wmma_f32_16x16x32_bf16(false, a, false, b, (short)0, c, false, false);
}

// ---------------------------------------------------------------------------
// gfx1250 async copy: global -> LDS, 16B per lane, tracked by ASYNCcnt.
// (inline asm per CDNA5 ISA; portable across toolchains)
// LDS operand = wave-relative LDS byte address = low 32 bits of generic ptr.
// ---------------------------------------------------------------------------
static __device__ __forceinline__ void async_b128_to_lds(const void* gptr, void* lds) {
  asm volatile("global_load_async_to_lds_b128 %0, %1, off"
               :: "v"((unsigned int)(uintptr_t)lds),
                  "v"((unsigned long long)(uintptr_t)gptr)
               : "memory");
}
static __device__ __forceinline__ void wait_async_all() {
  asm volatile("s_wait_asynccnt 0x0" ::: "memory");
}

// ---------------------------------------------------------------------------
// WMMA fragment loads (bf16 16x16x32), per ISA VGPR layout tables.
// A: row-major [M][K].  lane m = lane&15, K subset {b..b+7, b+16..b+23}, b=(lane>>4)*8
// B: stored transposed [N][K]. lane n = lane&15, K subset {b..b+7, b+16..b+23}, b=(lane>>4)*8
// ---------------------------------------------------------------------------
static __device__ __forceinline__ v16bf load_a_frag(const bf16_t* __restrict__ A, int lda,
                                                    int row0, int k0, int lane)
{
  const int m  = lane & 15;
  const int ks = (lane >> 4) << 3;
  const bf16_t* p = A + (size_t)(row0 + m) * lda + k0 + ks;
  union { v16bf v; v8bf h[2]; } u;
  u.h[0] = *(const v8bf*)(p);
  u.h[1] = *(const v8bf*)(p + 16);
  return u.v;
}

static __device__ __forceinline__ v16bf load_b_frag(const bf16_t* __restrict__ Bt, int ldb,
                                                    int n0, int k0, int lane)
{
  const int n  = lane & 15;
  const int ks = (lane >> 4) << 3;
  const bf16_t* p = Bt + (size_t)(n0 + n) * ldb + k0 + ks;
  union { v16bf v; v8bf h[2]; } u;
  u.h[0] = *(const v8bf*)(p);
  u.h[1] = *(const v8bf*)(p + 16);
  return u.v;
}

// ---------------------------------------------------------------------------
// Weight convert + transpose: W fp32 [K][N] row-major -> Wt bf16 [N][K]
// ---------------------------------------------------------------------------
__global__ __launch_bounds__(256) void wcvt_t_kernel(const float* __restrict__ W,
                                                     bf16_t* __restrict__ Wt,
                                                     int K, int N)
{
  __shared__ float tile[32][33];
  const int tx = threadIdx.x;          // 0..31
  const int ty = threadIdx.y;          // 0..7
  const int n0 = blockIdx.x * 32;
  const int k0 = blockIdx.y * 32;
  #pragma unroll
  for (int j = 0; j < 4; ++j)
    tile[ty + 8*j][tx] = W[(size_t)(k0 + ty + 8*j) * N + (n0 + tx)];
  __syncthreads();
  #pragma unroll
  for (int j = 0; j < 4; ++j)
    Wt[(size_t)(n0 + ty + 8*j) * K + (k0 + tx)] = (bf16_t)tile[tx][ty + 8*j];
}

// ---------------------------------------------------------------------------
// RMSNorm: per row of [rows, D]; writes fp32 (optional) and bf16 outputs
// ---------------------------------------------------------------------------
__global__ __launch_bounds__(256) void rmsnorm_kernel(const float* __restrict__ X,
                                                      const float* __restrict__ w,
                                                      float* __restrict__ outf,
                                                      bf16_t* __restrict__ outb,
                                                      int D)
{
  const int row = blockIdx.x;
  const int t = threadIdx.x;
  const float* x = X + (size_t)row * D;
  float s = 0.f;
  for (int c = t; c < D; c += 256) { float v = x[c]; s += v * v; }
  __shared__ float red[256];
  red[t] = s; __syncthreads();
  #pragma unroll
  for (int off = 128; off > 0; off >>= 1) {
    if (t < off) red[t] += red[t + off];
    __syncthreads();
  }
  const float scale = rsqrtf(red[0] / (float)D + 1e-6f);
  for (int c = t; c < D; c += 256) {
    float v = x[c] * scale * w[c];
    if (outf) outf[(size_t)row * D + c] = v;
    outb[(size_t)row * D + c] = (bf16_t)v;
  }
}

// ---------------------------------------------------------------------------
// GEMM with async-LDS A staging: C[M,N] = A_bf16[M,K] x Bt_bf16[N,K]
// Block = 256 threads = 8 waves (2 M x 4 N); wave tile 16*MI x 16*NI.
// A tile (32*MI x 32) double-buffered in LDS via global_load_async_to_lds_b128;
// B fragments stream from global (L2-resident weights).
// EPI: 0 = store fp32, 1 = silu fp32, 2 = +residual fp32
// ---------------------------------------------------------------------------
template<int MI, int NI, int EPI>
__global__ __launch_bounds__(256) void gemm_bf16_kernel(const bf16_t* __restrict__ A,
                                                        const bf16_t* __restrict__ Bt,
                                                        const float* __restrict__ res,
                                                        float* __restrict__ C,
                                                        int M, int N, int K)
{
  const int t = threadIdx.x;
  const int lane = t & 31;
  const int wave = t >> 5;
  const int wm = wave & 1;
  const int wn = wave >> 1;
  const int bm0 = blockIdx.y * (32 * MI);
  const int m0l = wm * (16 * MI);                 // local row base inside LDS tile
  const int n0  = blockIdx.x * (64 * NI) + wn * (16 * NI);

  __shared__ bf16_t Asm[2][32 * MI * 32];         // [buf][rows][32]

  const int arow  = t >> 1;                       // row within tile
  const int ahalf = (t & 1) * 16;                 // 16-elem (32B) half of the row

  v8f acc[MI][NI] = {};

  // preload tile 0
  if (arow < 32 * MI)
    async_b128_to_lds(A + (size_t)(bm0 + arow) * K + ahalf, &Asm[0][arow * 32 + ahalf]);

  const int KT = K / 32;
  for (int kt = 0; kt < KT; ++kt) {
    wait_async_all();
    __syncthreads();                               // current tile visible to all waves
    if (kt + 1 < KT && arow < 32 * MI)
      async_b128_to_lds(A + (size_t)(bm0 + arow) * K + (kt + 1) * 32 + ahalf,
                        &Asm[(kt + 1) & 1][arow * 32 + ahalf]);

    const bf16_t* At = &Asm[kt & 1][0];
    const int k0 = kt * 32;
    if (k0 + 64 < K)
      __builtin_prefetch(Bt + (size_t)(n0 + (lane & 15)) * K + k0 + 64, 0, 3);

    v16bf af[MI];
    v16bf bfr[NI];
    #pragma unroll
    for (int i = 0; i < MI; ++i) af[i]  = load_a_frag(At, 32, m0l + 16 * i, 0, lane);
    #pragma unroll
    for (int j = 0; j < NI; ++j) bfr[j] = load_b_frag(Bt, K, n0 + 16 * j, k0, lane);
    #pragma unroll
    for (int i = 0; i < MI; ++i)
      #pragma unroll
      for (int j = 0; j < NI; ++j)
        acc[i][j] = wmma_bf16(af[i], bfr[j], acc[i][j]);
    __syncthreads();                               // all reads done before overwrite
  }

  const int ccol = lane & 15;
  const int rofs = (lane >> 4) * 8;
  const int m0 = bm0 + m0l;
  #pragma unroll
  for (int i = 0; i < MI; ++i)
    #pragma unroll
    for (int j = 0; j < NI; ++j)
      #pragma unroll
      for (int r = 0; r < 8; ++r) {
        const size_t idx = (size_t)(m0 + 16 * i + rofs + r) * N + (n0 + 16 * j + ccol);
        float v = acc[i][j][r];
        if (EPI == 1) v = silu_f(v);
        else if (EPI == 2) v = v + res[idx];
        C[idx] = v;
      }
}

// ---------------------------------------------------------------------------
// Fused dual GEMM for SwiGLU (async-LDS A staging):
//   G_bf16 = bf16( silu(A@Wg) * (A@Wu) )
// ---------------------------------------------------------------------------
template<int MI, int NI>
__global__ __launch_bounds__(256) void gemm_dual_swiglu_kernel(const bf16_t* __restrict__ A,
                                                               const bf16_t* __restrict__ Bg,
                                                               const bf16_t* __restrict__ Bu,
                                                               bf16_t* __restrict__ G,
                                                               int M, int N, int K)
{
  const int t = threadIdx.x;
  const int lane = t & 31;
  const int wave = t >> 5;
  const int wm = wave & 1;
  const int wn = wave >> 1;
  const int bm0 = blockIdx.y * (32 * MI);
  const int m0l = wm * (16 * MI);
  const int n0  = blockIdx.x * (64 * NI) + wn * (16 * NI);

  __shared__ bf16_t Asm[2][32 * MI * 32];

  const int arow  = t >> 1;
  const int ahalf = (t & 1) * 16;

  v8f accg[MI][NI] = {};
  v8f accu[MI][NI] = {};

  if (arow < 32 * MI)
    async_b128_to_lds(A + (size_t)(bm0 + arow) * K + ahalf, &Asm[0][arow * 32 + ahalf]);

  const int KT = K / 32;
  for (int kt = 0; kt < KT; ++kt) {
    wait_async_all();
    __syncthreads();
    if (kt + 1 < KT && arow < 32 * MI)
      async_b128_to_lds(A + (size_t)(bm0 + arow) * K + (kt + 1) * 32 + ahalf,
                        &Asm[(kt + 1) & 1][arow * 32 + ahalf]);

    const bf16_t* At = &Asm[kt & 1][0];
    const int k0 = kt * 32;
    if (k0 + 64 < K) {
      __builtin_prefetch(Bg + (size_t)(n0 + (lane & 15)) * K + k0 + 64, 0, 3);
      __builtin_prefetch(Bu + (size_t)(n0 + (lane & 15)) * K + k0 + 64, 0, 3);
    }

    v16bf af[MI];
    v16bf bg[NI], bu[NI];
    #pragma unroll
    for (int i = 0; i < MI; ++i) af[i] = load_a_frag(At, 32, m0l + 16 * i, 0, lane);
    #pragma unroll
    for (int j = 0; j < NI; ++j) { bg[j] = load_b_frag(Bg, K, n0 + 16 * j, k0, lane);
                                   bu[j] = load_b_frag(Bu, K, n0 + 16 * j, k0, lane); }
    #pragma unroll
    for (int i = 0; i < MI; ++i)
      #pragma unroll
      for (int j = 0; j < NI; ++j) {
        accg[i][j] = wmma_bf16(af[i], bg[j], accg[i][j]);
        accu[i][j] = wmma_bf16(af[i], bu[j], accu[i][j]);
      }
    __syncthreads();
  }

  const int ccol = lane & 15;
  const int rofs = (lane >> 4) * 8;
  const int m0 = bm0 + m0l;
  #pragma unroll
  for (int i = 0; i < MI; ++i)
    #pragma unroll
    for (int j = 0; j < NI; ++j)
      #pragma unroll
      for (int r = 0; r < 8; ++r) {
        const size_t idx = (size_t)(m0 + 16 * i + rofs + r) * N + (n0 + 16 * j + ccol);
        G[idx] = (bf16_t)(silu_f(accg[i][j][r]) * accu[i][j][r]);
      }
}

// ---------------------------------------------------------------------------
// Per-(token,head): l2norm q,k in-place; beta = sigmoid(x . Wb[:,h])
// ---------------------------------------------------------------------------
__global__ __launch_bounds__(64) void qknorm_beta_kernel(float* __restrict__ q,
                                                         float* __restrict__ k,
                                                         const float* __restrict__ x,
                                                         const float* __restrict__ Wb,
                                                         float* __restrict__ beta,
                                                         int D, int H)
{
  const int blk = blockIdx.x;
  const int row = blk / H;
  const int h   = blk % H;
  const int t   = threadIdx.x;
  const size_t base = (size_t)row * D + h * 64;
  __shared__ float red[64];

  float qv = q[base + t];
  red[t] = qv * qv; __syncthreads();
  #pragma unroll
  for (int off = 32; off > 0; off >>= 1) { if (t < off) red[t] += red[t + off]; __syncthreads(); }
  const float qs = rsqrtf(red[0] + 1e-6f); __syncthreads();

  float kv = k[base + t];
  red[t] = kv * kv; __syncthreads();
  #pragma unroll
  for (int off = 32; off > 0; off >>= 1) { if (t < off) red[t] += red[t + off]; __syncthreads(); }
  const float ks = rsqrtf(red[0] + 1e-6f); __syncthreads();

  q[base + t] = qv * qs;
  k[base + t] = kv * ks;

  float p = 0.f;
  for (int d = t; d < D; d += 64)
    p += x[(size_t)row * D + d] * Wb[(size_t)d * H + h];
  red[t] = p; __syncthreads();
  #pragma unroll
  for (int off = 32; off > 0; off >>= 1) { if (t < off) red[t] += red[t + off]; __syncthreads(); }
  if (t == 0) beta[(size_t)row * H + h] = sigmoid_f(red[0]);
}

// ---------------------------------------------------------------------------
// Chunked delta rule, stage 1 (fully parallel over (b,h,chunk)):
//   A   = K K^T (WMMA, raw; row-scaled by beta on use)
//   D   = tril_incl(Q K^T) (WMMA -> global bf16)
//   solve (I + strict_tril(diag(b)KK^T)) [W|U0] = diag(b)[K|V]  (forward subst.)
// ---------------------------------------------------------------------------
__global__ __launch_bounds__(256) void delta_prep_kernel(const float* __restrict__ q,
                                                         const float* __restrict__ k,
                                                         const float* __restrict__ v,
                                                         const float* __restrict__ beta,
                                                         bf16_t* __restrict__ Wc,
                                                         float*  __restrict__ U0t,
                                                         bf16_t* __restrict__ Qc,
                                                         bf16_t* __restrict__ Dc,
                                                         bf16_t* __restrict__ Ktc,
                                                         int Sq, int D, int H, int NC)
{
  const int g = blockIdx.x;           // ((b*H + h)*NC + c)
  const int c = g % NC;
  const int h = (g / NC) % H;
  const int b = g / (NC * H);
  const int t = threadIdx.x;
  const int lane = t & 31;
  const int wave = t >> 5;
  const int wm = wave & 1, wn = wave >> 1;
  const int ccol = lane & 15;
  const int rofs = (lane >> 4) * 8;
  const size_t gbase = (size_t)g * 4096;
  const size_t rbase = ((size_t)b * Sq + (size_t)c * 64) * D + h * 64;

  __shared__ float  X[64 * 128];      // solve workspace (aliases Qb early)
  __shared__ float  Al[64 * 64];      // raw K K^T
  __shared__ bf16_t Kb[64 * 64];
  __shared__ float  bl[64];
  bf16_t* Qb = (bf16_t*)X;

  if (t < 64) bl[t] = beta[((size_t)b * Sq + (size_t)c * 64 + t) * H + h];
  for (int i = t; i < 4096; i += 256) {
    const int row = i >> 6, d = i & 63;
    Kb[i] = (bf16_t)k[rbase + (size_t)row * D + d];
    Qb[i] = (bf16_t)q[rbase + (size_t)row * D + d];
  }
  __syncthreads();

  #pragma unroll
  for (int i = 0; i < 2; ++i) {
    const int m0 = wm * 32 + i * 16;
    const int n0 = wn * 16;
    v8f accA = {};
    v8f accD = {};
    #pragma unroll
    for (int k0 = 0; k0 < 64; k0 += 32) {
      v16bf bb = load_b_frag(Kb, 64, n0, k0, lane);
      v16bf aK = load_a_frag(Kb, 64, m0, k0, lane);
      v16bf aQ = load_a_frag(Qb, 64, m0, k0, lane);
      accA = wmma_bf16(aK, bb, accA);
      accD = wmma_bf16(aQ, bb, accD);
    }
    #pragma unroll
    for (int r = 0; r < 8; ++r) {
      const int row = m0 + rofs + r, col = n0 + ccol;
      Al[row * 64 + col] = accA[r];
      Dc[gbase + row * 64 + col] = (bf16_t)(row >= col ? accD[r] : 0.f);
    }
  }
  __syncthreads();

  for (int i = t; i < 4096; i += 256) {
    const int row = i >> 6, d = i & 63;
    const float be = bl[row];
    X[row * 128 + d]      = be * k[rbase + (size_t)row * D + d];
    X[row * 128 + 64 + d] = be * v[rbase + (size_t)row * D + d];
    Qc[gbase + i] = (bf16_t)q[rbase + (size_t)row * D + d];
  }
  __syncthreads();

  const int cx = t & 127;
  const int rg = t >> 7;
  for (int i = 0; i < 63; ++i) {
    const float xi = X[i * 128 + cx];
    for (int r = i + 1 + rg; r < 64; r += 2)
      X[r * 128 + cx] -= bl[r] * Al[r * 64 + i] * xi;
    __syncthreads();
  }

  for (int idx = t; idx < 4096; idx += 256) {
    const int row = idx >> 6, col = idx & 63;
    Wc[gbase + idx]  = (bf16_t)X[row * 128 + col];
    U0t[gbase + idx] = X[col * 128 + 64 + row];   // U0^T: (v,i) = X[i][64+v]
    Ktc[gbase + idx] = Kb[col * 64 + row];        // K^T:  (d,i) = K[i][d]
  }
}

// ---------------------------------------------------------------------------
// Chunked delta rule, stage 2 (sequential over chunks, one block per (b,h)):
// state kept transposed St = S^T [dv][dk] in LDS (fp32 + bf16 shadow).
// ---------------------------------------------------------------------------
__global__ __launch_bounds__(256) void delta_scan_kernel(const bf16_t* __restrict__ Wc,
                                                         const float*  __restrict__ U0t,
                                                         const bf16_t* __restrict__ Qc,
                                                         const bf16_t* __restrict__ Dc,
                                                         const bf16_t* __restrict__ Ktc,
                                                         float* __restrict__ o,
                                                         int Sq, int D, int H, int NC)
{
  const int bh = blockIdx.x;          // b*H + h
  const int b = bh / H, h = bh % H;
  const int t = threadIdx.x;
  const int lane = t & 31;
  const int wave = t >> 5;
  const int wm = wave & 1, wn = wave >> 1;
  const int ccol = lane & 15;
  const int rofs = (lane >> 4) * 8;

  __shared__ float  Sf[64 * 64];      // S^T fp32
  __shared__ bf16_t Sb[64 * 64];      // S^T bf16
  __shared__ bf16_t Ub[64 * 64];      // U^T bf16

  for (int i = t; i < 4096; i += 256) Sf[i] = 0.f;
  __syncthreads();

  for (int c = 0; c < NC; ++c) {
    const size_t gbase = ((size_t)bh * NC + c) * 4096;

    for (int i = t; i < 4096; i += 256) Sb[i] = (bf16_t)Sf[i];
    __syncthreads();

    // U^T = U0^T - St_bf @ W  (acc seeded with -U0^T, result negated on store)
    #pragma unroll
    for (int i = 0; i < 2; ++i) {
      const int m0 = wm * 32 + i * 16;
      const int n0 = wn * 16;
      v8f acc;
      #pragma unroll
      for (int r = 0; r < 8; ++r)
        acc[r] = -U0t[gbase + (size_t)(m0 + rofs + r) * 64 + (n0 + ccol)];
      #pragma unroll
      for (int k0 = 0; k0 < 64; k0 += 32) {
        v16bf a  = load_a_frag(Sb, 64, m0, k0, lane);
        v16bf bb = load_b_frag(Wc + gbase, 64, n0, k0, lane);
        acc = wmma_bf16(a, bb, acc);
      }
      #pragma unroll
      for (int r = 0; r < 8; ++r)
        Ub[(m0 + rofs + r) * 64 + (n0 + ccol)] = (bf16_t)(-acc[r]);
    }
    __syncthreads();

    // O = Q @ S0 + D @ U   (pre-update state)
    #pragma unroll
    for (int i = 0; i < 2; ++i) {
      const int m0 = wm * 32 + i * 16;
      const int n0 = wn * 16;
      v8f acc = {};
      #pragma unroll
      for (int k0 = 0; k0 < 64; k0 += 32) {
        v16bf a  = load_a_frag(Qc + gbase, 64, m0, k0, lane);
        v16bf bb = load_b_frag(Sb, 64, n0, k0, lane);
        acc = wmma_bf16(a, bb, acc);
      }
      #pragma unroll
      for (int k0 = 0; k0 < 64; k0 += 32) {
        v16bf a  = load_a_frag(Dc + gbase, 64, m0, k0, lane);
        v16bf bb = load_b_frag(Ub, 64, n0, k0, lane);
        acc = wmma_bf16(a, bb, acc);
      }
      #pragma unroll
      for (int r = 0; r < 8; ++r) {
        const int row = m0 + rofs + r;
        const int col = n0 + ccol;
        o[((size_t)b * Sq + (size_t)c * 64 + row) * D + h * 64 + col] = acc[r];
      }
    }

    // St += U^T_bf @ K^T  (each wave RMWs only its own tiles)
    #pragma unroll
    for (int i = 0; i < 2; ++i) {
      const int m0 = wm * 32 + i * 16;
      const int n0 = wn * 16;
      v8f acc;
      #pragma unroll
      for (int r = 0; r < 8; ++r)
        acc[r] = Sf[(m0 + rofs + r) * 64 + (n0 + ccol)];
      #pragma unroll
      for (int k0 = 0; k0 < 64; k0 += 32) {
        v16bf a  = load_a_frag(Ub, 64, m0, k0, lane);
        v16bf bb = load_b_frag(Ktc + gbase, 64, n0, k0, lane);
        acc = wmma_bf16(a, bb, acc);
      }
      #pragma unroll
      for (int r = 0; r < 8; ++r)
        Sf[(m0 + rofs + r) * 64 + (n0 + ccol)] = acc[r];
    }
    __syncthreads();
  }
}

// ---------------------------------------------------------------------------
// fp32 -> bf16 convert (vec4)
// ---------------------------------------------------------------------------
__global__ __launch_bounds__(256) void f32_to_bf16_kernel(const float* __restrict__ in,
                                                          bf16_t* __restrict__ out,
                                                          size_t n)
{
  const size_t i = ((size_t)blockIdx.x * 256 + threadIdx.x) * 4;
  if (i + 3 < n) {
    float4 f = *(const float4*)(in + i);
    out[i + 0] = (bf16_t)f.x;
    out[i + 1] = (bf16_t)f.y;
    out[i + 2] = (bf16_t)f.z;
    out[i + 3] = (bf16_t)f.w;
  }
}

// ---------------------------------------------------------------------------
// Launcher
// ---------------------------------------------------------------------------
extern "C" void kernel_launch(void* const* d_in, const int* in_sizes, int n_in,
                              void* d_out, int out_size, void* d_ws, size_t ws_size,
                              hipStream_t stream)
{
  (void)in_sizes; (void)n_in; (void)out_size; (void)ws_size;
  constexpr int B = 2, S = 2048, D = 1024, H = 16, F = 4096;
  constexpr int M = B * S;              // 4096 token rows
  constexpr int NC = S / 64;            // 32 chunks per sequence
  constexpr int G = B * H * NC;         // 1024 chunk instances

  const float* hidden = (const float*)d_in[0];
  const float* Wq  = (const float*)d_in[1];
  const float* Wk  = (const float*)d_in[2];
  const float* Wv  = (const float*)d_in[3];
  const float* Wb  = (const float*)d_in[4];
  const float* Wo  = (const float*)d_in[5];
  const float* ln1 = (const float*)d_in[6];
  const float* ln2 = (const float*)d_in[7];
  const float* Wg  = (const float*)d_in[8];
  const float* Wu  = (const float*)d_in[9];
  const float* Wd  = (const float*)d_in[10];
  float* out = (float*)d_out;

  char* ws = (char*)d_ws;
  size_t off = 0;
  auto alloc = [&](size_t bytes) -> char* {
    char* p = ws + off;
    off += (bytes + 255) & ~(size_t)255;
    return p;
  };
  bf16_t* Wq_t  = (bf16_t*)alloc((size_t)D * D * 2);
  bf16_t* Wk_t  = (bf16_t*)alloc((size_t)D * D * 2);
  bf16_t* Wv_t  = (bf16_t*)alloc((size_t)D * D * 2);
  bf16_t* Wo_t  = (bf16_t*)alloc((size_t)D * D * 2);
  bf16_t* Wg_t  = (bf16_t*)alloc((size_t)D * F * 2);
  bf16_t* Wu_t  = (bf16_t*)alloc((size_t)D * F * 2);
  bf16_t* Wd_t  = (bf16_t*)alloc((size_t)F * D * 2);
  float*  x_f32 = (float*)alloc((size_t)M * D * 4);
  bf16_t* x_b16 = (bf16_t*)alloc((size_t)M * D * 2);
  float*  qb    = (float*)alloc((size_t)M * D * 4);
  float*  kb    = (float*)alloc((size_t)M * D * 4);
  float*  vb    = (float*)alloc((size_t)M * D * 4);
  float*  betab = (float*)alloc((size_t)M * H * 4);
  float*  o_f32 = (float*)alloc((size_t)M * D * 4);
  bf16_t* o_b16 = (bf16_t*)alloc((size_t)M * D * 2);
  float*  h_f32 = (float*)alloc((size_t)M * D * 4);
  bf16_t* y_b16 = (bf16_t*)alloc((size_t)M * D * 2);
  bf16_t* g_b16 = (bf16_t*)alloc((size_t)M * F * 2);
  bf16_t* Wc    = (bf16_t*)alloc((size_t)G * 4096 * 2);
  float*  U0t   = (float*) alloc((size_t)G * 4096 * 4);
  bf16_t* Qc    = (bf16_t*)alloc((size_t)G * 4096 * 2);
  bf16_t* Dc    = (bf16_t*)alloc((size_t)G * 4096 * 2);
  bf16_t* Ktc   = (bf16_t*)alloc((size_t)G * 4096 * 2);

  const dim3 tb(32, 8);
  wcvt_t_kernel<<<dim3(D / 32, D / 32), tb, 0, stream>>>(Wq, Wq_t, D, D);
  wcvt_t_kernel<<<dim3(D / 32, D / 32), tb, 0, stream>>>(Wk, Wk_t, D, D);
  wcvt_t_kernel<<<dim3(D / 32, D / 32), tb, 0, stream>>>(Wv, Wv_t, D, D);
  wcvt_t_kernel<<<dim3(D / 32, D / 32), tb, 0, stream>>>(Wo, Wo_t, D, D);
  wcvt_t_kernel<<<dim3(F / 32, D / 32), tb, 0, stream>>>(Wg, Wg_t, D, F);
  wcvt_t_kernel<<<dim3(F / 32, D / 32), tb, 0, stream>>>(Wu, Wu_t, D, F);
  wcvt_t_kernel<<<dim3(D / 32, F / 32), tb, 0, stream>>>(Wd, Wd_t, F, D);

  rmsnorm_kernel<<<M, 256, 0, stream>>>(hidden, ln1, x_f32, x_b16, D);

  gemm_bf16_kernel<4, 2, 1><<<dim3(D / 128, M / 128), 256, 0, stream>>>(x_b16, Wq_t, nullptr, qb, M, D, D);
  gemm_bf16_kernel<4, 2, 1><<<dim3(D / 128, M / 128), 256, 0, stream>>>(x_b16, Wk_t, nullptr, kb, M, D, D);
  gemm_bf16_kernel<4, 2, 1><<<dim3(D / 128, M / 128), 256, 0, stream>>>(x_b16, Wv_t, nullptr, vb, M, D, D);

  qknorm_beta_kernel<<<M * H, 64, 0, stream>>>(qb, kb, x_f32, Wb, betab, D, H);

  delta_prep_kernel<<<G, 256, 0, stream>>>(qb, kb, vb, betab, Wc, U0t, Qc, Dc, Ktc, S, D, H, NC);
  delta_scan_kernel<<<B * H, 256, 0, stream>>>(Wc, U0t, Qc, Dc, Ktc, o_f32, S, D, H, NC);

  f32_to_bf16_kernel<<<(int)(((size_t)M * D / 4 + 255) / 256), 256, 0, stream>>>(o_f32, o_b16, (size_t)M * D);
  gemm_bf16_kernel<4, 2, 2><<<dim3(D / 128, M / 128), 256, 0, stream>>>(o_b16, Wo_t, hidden, h_f32, M, D, D);

  rmsnorm_kernel<<<M, 256, 0, stream>>>(h_f32, ln2, nullptr, y_b16, D);

  gemm_dual_swiglu_kernel<2, 2><<<dim3(F / 128, M / 64), 256, 0, stream>>>(y_b16, Wg_t, Wu_t, g_b16, M, F, D);

  gemm_bf16_kernel<4, 2, 2><<<dim3(D / 128, M / 128), 256, 0, stream>>>(g_b16, Wd_t, h_f32, out, M, D, F);
}
